// GINLayer_27951647162471
// MI455X (gfx1250) — matrix operand order repaired and checked by reading, not compiled, not run on hardware
//
#include <hip/hip_runtime.h>

// ---------------------------------------------------------------------------
// GIN layer on MI455X (gfx1250, wave32, WMMA).
//   P[16384 x 272] = bf16(adj) @ Xext          (Xext = [X | 1 | 0...], WMMA bf16)
//   out = (1+eps)*X + (1/(P[:,256]+1)) * (P[:,:256] + X)      (identity folded in)
//   h1  = relu(out_bf16 @ W1 + b1)   (WMMA bf16)
//   h2  = h1 @ W2 + b2               (WMMA bf16, fp32 out)
//   BN  : deterministic 2-stage column reduction, then normalize.
// ---------------------------------------------------------------------------

#define NN   16384   // nodes (M and K of the big GEMM)
#define DIM  256     // feature dim
#define NE   272     // DIM + 16 (ones column tile for row sums)

typedef __attribute__((ext_vector_type(16))) __bf16 v16bf;
typedef __attribute__((ext_vector_type(8)))  float  v8f;

union BF16Frag {
    v16bf bf;
    int4  q[2];
    int   i[8];
};

// ---- fp32 -> bf16 (RNE).  Prefer HW v_cvt_pk_bf16_f32 when available. ----
__device__ __forceinline__ unsigned int f2bf_bits(float f) {
    unsigned int u = __float_as_uint(f);
    return (u + 0x7FFFu + ((u >> 16) & 1u)) >> 16;
}
#if __has_builtin(__builtin_amdgcn_cvt_pk_bf16_f32)
typedef __attribute__((ext_vector_type(2))) __bf16 v2bf;
__device__ __forceinline__ int pack2bf(float lo, float hi) {
    v2bf v = __builtin_amdgcn_cvt_pk_bf16_f32(lo, hi);
    int r;
    __builtin_memcpy(&r, &v, sizeof(r));
    return r;
}
#else
__device__ __forceinline__ int pack2bf(float lo, float hi) {
    return (int)(f2bf_bits(lo) | (f2bf_bits(hi) << 16));
}
#endif
__device__ __forceinline__ unsigned short f2bf(float f) {
    return (unsigned short)f2bf_bits(f);
}

// ---- CDNA5 async global->LDS staging (ASYNCcnt-tracked), if toolchain has it.
#if __has_builtin(__builtin_amdgcn_global_load_async_to_lds_b128) && \
    __has_builtin(__builtin_amdgcn_s_wait_asynccnt)
#define GIN_ASYNC 1
typedef __attribute__((__vector_size__(4 * sizeof(int)))) int nvec_int4;  // native vec
typedef __attribute__((address_space(1))) nvec_int4 as1_int4;   // global
typedef __attribute__((address_space(3))) nvec_int4 as3_int4;   // LDS
__device__ __forceinline__ void async_b128(const void* g, void* l) {
    __builtin_amdgcn_global_load_async_to_lds_b128(
        (as1_int4*)g, (as3_int4*)l, 0, 0);
}
#else
#define GIN_ASYNC 0
#endif

// --------------------------------------------------------------------------
// Pack features -> bf16 WMMA-B layout with an extra ones column.
// Xp[((k>>4)*NE + n)*16 + (k&15)]  : lane fragment = 16 contiguous bf16.
// --------------------------------------------------------------------------
__global__ void pack_features_kernel(const float* __restrict__ X,
                                     unsigned short* __restrict__ xp) {
    const int k = blockIdx.x;      // 0..NN-1
    const int n = threadIdx.x;     // 0..255
    const float v = X[(size_t)k * DIM + n];
    xp[((size_t)(k >> 4) * NE + n) * 16 + (k & 15)] = f2bf(v);
    if (n < 16) {
        const int n2 = DIM + n;
        const float v2 = (n == 0) ? 1.0f : 0.0f;   // ones column -> row sums
        xp[((size_t)(k >> 4) * NE + n2) * 16 + (k & 15)] = f2bf(v2);
    }
}

// W[256x256] -> bf16 WMMA-B layout.
__global__ void pack_weight_kernel(const float* __restrict__ W,
                                   unsigned short* __restrict__ wp) {
    const int k = blockIdx.x;      // 0..255
    const int n = threadIdx.x;     // 0..255
    wp[((size_t)(k >> 4) * DIM + n) * 16 + (k & 15)] = f2bf(W[(size_t)k * DIM + n]);
}

// --------------------------------------------------------------------------
// P = bf16(adj) @ Xext.  128 blocks x 256 thr (8 waves).  Block = 128 rows x 272 cols.
// adj read once from HBM as fp32, converted to bf16 A-fragments in registers.
// B slab (32 x 272 bf16 = 17408 B) staged in LDS; double-buffered async copy
// (GLOBAL_LOAD_ASYNC_TO_LDS_B128) overlapped with the 17 WMMAs when available.
// --------------------------------------------------------------------------
__global__ __launch_bounds__(256) void gemm_adj_kernel(
        const float* __restrict__ adj,
        const char*  __restrict__ xp,     // packed Xext (bf16)
        float*       __restrict__ P) {
#if GIN_ASYNC
    __shared__ int4 ldsB[2][1088];        // 2 x 17408 bytes (double buffer)
#else
    __shared__ int4 ldsB[1][1088];
#endif

    const int tid  = threadIdx.x;
    const int lane = tid & 31;
    const int wave = tid >> 5;
    const int ln16 = lane & 15;
    const int hi   = lane >> 4;           // 0 or 1 (half-wave)
    const int mbase = blockIdx.x * 128 + wave * 16;
    const int arow  = mbase + ln16;
    const float* aptr = adj + (size_t)arow * NN + hi * 16;

    v8f zero = {0.f, 0.f, 0.f, 0.f, 0.f, 0.f, 0.f, 0.f};
    v8f acc[17];
#pragma unroll
    for (int t = 0; t < 17; ++t) acc[t] = zero;

#if GIN_ASYNC
    // prologue: async-copy slab 0 into buffer 0
    for (int i = tid; i < 1088; i += 256)
        async_b128(xp + (size_t)i * 16, (char*)&ldsB[0][0] + (size_t)i * 16);
    __builtin_amdgcn_s_wait_asynccnt(0);
    __syncthreads();
#endif

    for (int kb = 0; kb < NN / 32; ++kb) {
#if GIN_ASYNC
        const int cur = kb & 1;
        if (kb + 1 < NN / 32) {           // overlap next slab copy with compute
            const size_t nb = (size_t)(kb + 1) * 17408;
            for (int i = tid; i < 1088; i += 256)
                async_b128(xp + nb + (size_t)i * 16,
                           (char*)&ldsB[cur ^ 1][0] + (size_t)i * 16);
        }
#else
        const int cur = 0;
        const int4* src = (const int4*)(xp + (size_t)kb * 17408);
        __syncthreads();
        for (int i = tid; i < 1088; i += 256) ldsB[0][i] = src[i];
        __syncthreads();
#endif

        if (kb + 1 < NN / 32)
            __builtin_prefetch((const void*)(aptr + (size_t)(kb + 1) * 32), 0, 1);

        // ---- A fragment: 16 fp32 -> 16 bf16, contiguous k = hi*16 + s (same as B) ----
        const float4* ap = (const float4*)(aptr + (size_t)kb * 32);
        float4 a0 = ap[0], a1 = ap[1];        // K = hi*16 + 0..7
        float4 a2 = ap[2], a3 = ap[3];        // K = hi*16 + 8..15
        BF16Frag A;
        A.i[0] = pack2bf(a0.x, a0.y); A.i[1] = pack2bf(a0.z, a0.w);
        A.i[2] = pack2bf(a1.x, a1.y); A.i[3] = pack2bf(a1.z, a1.w);
        A.i[4] = pack2bf(a2.x, a2.y); A.i[5] = pack2bf(a2.z, a2.w);
        A.i[6] = pack2bf(a3.x, a3.y); A.i[7] = pack2bf(a3.z, a3.w);

        const char* lbase = (const char*)&ldsB[cur][0] + (size_t)(hi * NE + ln16) * 32;
#pragma unroll
        for (int t = 0; t < 17; ++t) {
            BF16Frag B;
            const int4* bp = (const int4*)(lbase + (size_t)t * 16 * 32);
            B.q[0] = bp[0];
            B.q[1] = bp[1];
            acc[t] = __builtin_amdgcn_wmma_f32_16x16x32_bf16(
                false, A.bf, false, B.bf, (short)0, acc[t], false, false);
        }

#if GIN_ASYNC
        __builtin_amdgcn_s_wait_asynccnt(0);  // next slab fully in LDS
        __syncthreads();                      // all waves done reading cur
#endif
    }

    // ---- write C per ISA f32 16x16 layout ----
#pragma unroll
    for (int t = 0; t < 17; ++t) {
        const int col = t * 16 + ln16;
#pragma unroll
        for (int r = 0; r < 8; ++r) {
            const int row = mbase + hi * 8 + r;
            P[(size_t)row * NE + col] = acc[t][r];
        }
    }
}

// --------------------------------------------------------------------------
// out = (1+eps)*X + dinv*(P[:,:256] + X);  dinv = 1/(P[:,256]+1);  -> bf16
// --------------------------------------------------------------------------
__global__ void epilogue1_kernel(const float* __restrict__ X,
                                 const float* __restrict__ P,
                                 const float* __restrict__ epsp,
                                 unsigned short* __restrict__ outbf) {
    const int t   = blockIdx.x * 256 + threadIdx.x;   // NN*64 threads
    const int row = t >> 6;
    const int c   = (t & 63) << 2;
    const float rs   = P[(size_t)row * NE + DIM] + 1.0f;
    const float dinv = (rs != 0.f) ? 1.0f / rs : 0.0f;
    const float ope  = 1.0f + *epsp;
    const float4 x = *(const float4*)(X + (size_t)row * DIM + c);
    const float4 p = *(const float4*)(P + (size_t)row * NE + c);
    ushort4 o;
    o.x = f2bf(ope * x.x + dinv * (p.x + x.x));
    o.y = f2bf(ope * x.y + dinv * (p.y + x.y));
    o.z = f2bf(ope * x.z + dinv * (p.z + x.z));
    o.w = f2bf(ope * x.w + dinv * (p.w + x.w));
    *(ushort4*)(outbf + (size_t)row * DIM + c) = o;
}

// --------------------------------------------------------------------------
// h = act(Abf @ Wp + bias).  A bf16 row-major [NN][256], Wp packed B layout.
// 128 blocks x 256 thr; block = 128 rows x 256 cols; K = 256 (8 steps).
// --------------------------------------------------------------------------
template <bool RELU, bool OUTBF>
__global__ __launch_bounds__(256) void gemm_mlp_kernel(
        const unsigned short* __restrict__ Abf,
        const char*  __restrict__ wp,
        const float* __restrict__ bias,
        void*        __restrict__ outp) {
    __shared__ int4 ldsB[1024];           // 16 KB: one 32x256 bf16 W slab

    const int tid  = threadIdx.x;
    const int lane = tid & 31;
    const int wave = tid >> 5;
    const int ln16 = lane & 15;
    const int hi   = lane >> 4;
    const int mbase = blockIdx.x * 128 + wave * 16;
    const int arow  = mbase + ln16;

    v8f zero = {0.f, 0.f, 0.f, 0.f, 0.f, 0.f, 0.f, 0.f};
    v8f acc[16];
#pragma unroll
    for (int t = 0; t < 16; ++t) acc[t] = zero;

    for (int kb = 0; kb < DIM / 32; ++kb) {
        const int4* src = (const int4*)(wp + (size_t)kb * 16384);
        __syncthreads();
#pragma unroll
        for (int i = 0; i < 4; ++i) ldsB[tid + i * 256] = src[tid + i * 256];
        __syncthreads();

        BF16Frag A;   // A is already bf16 row-major: one contiguous 16-elem run
        const int4* ap = (const int4*)(Abf + (size_t)arow * DIM + kb * 32 + hi * 16);
        A.q[0] = ap[0];   // K = hi*16 + 0..7
        A.q[1] = ap[1];   // K = hi*16 + 8..15  (+16 bytes)

        const char* lbase = (const char*)ldsB + (size_t)(hi * DIM + ln16) * 32;
#pragma unroll
        for (int t = 0; t < 16; ++t) {
            BF16Frag B;
            const int4* bp = (const int4*)(lbase + (size_t)t * 16 * 32);
            B.q[0] = bp[0];
            B.q[1] = bp[1];
            acc[t] = __builtin_amdgcn_wmma_f32_16x16x32_bf16(
                false, A.bf, false, B.bf, (short)0, acc[t], false, false);
        }
    }

#pragma unroll
    for (int t = 0; t < 16; ++t) {
        const int col = t * 16 + ln16;
        const float b = bias[col];
#pragma unroll
        for (int r = 0; r < 8; ++r) {
            float v = acc[t][r] + b;
            if (RELU) v = (v > 0.f) ? v : 0.f;
            const int row = mbase + hi * 8 + r;
            if (OUTBF)
                ((unsigned short*)outp)[(size_t)row * DIM + col] = f2bf(v);
            else
                ((float*)outp)[(size_t)row * DIM + col] = v;
        }
    }
}

// --------------------------------------------------------------------------
// BatchNorm: deterministic two-stage column reduction + normalize.
// --------------------------------------------------------------------------
__global__ void bn_stats1_kernel(const float* __restrict__ h2,
                                 float* __restrict__ part) {
    const int b = blockIdx.x;     // 128 blocks of 128 rows
    const int c = threadIdx.x;    // 256 cols
    float s = 0.f, q = 0.f;
    const float* p = h2 + (size_t)b * 128 * DIM + c;
    for (int r = 0; r < 128; ++r) {
        const float v = p[(size_t)r * DIM];
        s += v;
        q += v * v;
    }
    part[(size_t)b * DIM + c] = s;
    part[(size_t)128 * DIM + (size_t)b * DIM + c] = q;
}

__global__ void bn_stats2_kernel(const float* __restrict__ part,
                                 const float* __restrict__ gamma,
                                 const float* __restrict__ beta,
                                 float* __restrict__ ss) {
    const int c = threadIdx.x;
    float s = 0.f, q = 0.f;
    for (int b = 0; b < 128; ++b) {
        s += part[(size_t)b * DIM + c];
        q += part[(size_t)128 * DIM + (size_t)b * DIM + c];
    }
    const float mean = s * (1.0f / (float)NN);
    const float var  = q * (1.0f / (float)NN) - mean * mean;
    const float sc   = gamma[c] * rsqrtf(var + 1e-5f);
    ss[c]        = sc;
    ss[DIM + c]  = beta[c] - mean * sc;
}

__global__ void bn_apply_kernel(const float* __restrict__ h2,
                                const float* __restrict__ ss,
                                float* __restrict__ out) {
    const size_t t   = (size_t)blockIdx.x * 256 + threadIdx.x;
    const size_t idx = t * 4;
    const int c = (int)(idx & (DIM - 1));
    const float4 v  = *(const float4*)(h2 + idx);
    const float4 sc = *(const float4*)(ss + c);
    const float4 sh = *(const float4*)(ss + DIM + c);
    float4 o;
    o.x = v.x * sc.x + sh.x;
    o.y = v.y * sc.y + sh.y;
    o.z = v.z * sc.z + sh.z;
    o.w = v.w * sc.w + sh.w;
    *(float4*)(out + idx) = o;
}

// --------------------------------------------------------------------------
// Host launcher.  ws layout (bytes), with region reuse:
//   [0, 17.8MB)  P          -> later reused for h2 (fp32)
//   [17.8, 26.7) Xp (bf16)  -> later reused for h1 (bf16)
//   [26.7, 35.1) out_bf16
//   then Wp1, Wp2, partials, scale/shift.   Total ~35.7 MB.
// --------------------------------------------------------------------------
extern "C" void kernel_launch(void* const* d_in, const int* in_sizes, int n_in,
                              void* d_out, int out_size, void* d_ws, size_t ws_size,
                              hipStream_t stream) {
    const float* adj      = (const float*)d_in[0];
    const float* features = (const float*)d_in[1];
    const float* epsilon  = (const float*)d_in[2];
    const float* W1       = (const float*)d_in[3];
    const float* b1       = (const float*)d_in[4];
    const float* W2       = (const float*)d_in[5];
    const float* b2       = (const float*)d_in[6];
    const float* gamma    = (const float*)d_in[7];
    const float* beta     = (const float*)d_in[8];

    char* ws = (char*)d_ws;
    const size_t off_P     = 0;                          // 16384*272*4 = 17825792
    const size_t off_h2    = 0;                          // reuses P
    const size_t off_Xp    = 17825792;                   // 16384*272*2 = 8912896
    const size_t off_h1    = 17825792;                   // reuses Xp
    const size_t off_outbf = off_Xp + 8912896;           // 26738688, 8388608 B
    const size_t off_Wp1   = off_outbf + 8388608;        // 35127296, 131072 B
    const size_t off_Wp2   = off_Wp1 + 131072;           // 131072 B
    const size_t off_part  = off_Wp2 + 131072;           // 128*256*2*4 = 262144 B
    const size_t off_ss    = off_part + 262144;          // 2048 B

    float*          P     = (float*)(ws + off_P);
    unsigned short* Xp    = (unsigned short*)(ws + off_Xp);
    unsigned short* outbf = (unsigned short*)(ws + off_outbf);
    unsigned short* Wp1   = (unsigned short*)(ws + off_Wp1);
    unsigned short* Wp2   = (unsigned short*)(ws + off_Wp2);
    unsigned short* h1    = (unsigned short*)(ws + off_h1);
    float*          h2    = (float*)(ws + off_h2);
    float*          part  = (float*)(ws + off_part);
    float*          ss    = (float*)(ws + off_ss);

    pack_features_kernel<<<NN, 256, 0, stream>>>(features, Xp);
    pack_weight_kernel<<<DIM, 256, 0, stream>>>(W1, Wp1);
    pack_weight_kernel<<<DIM, 256, 0, stream>>>(W2, Wp2);

    gemm_adj_kernel<<<NN / 128, 256, 0, stream>>>(adj, (const char*)Xp, P);

    epilogue1_kernel<<<NN * 64 / 256, 256, 0, stream>>>(features, P, epsilon, outbf);

    gemm_mlp_kernel<true,  true ><<<NN / 128, 256, 0, stream>>>(outbf, (const char*)Wp1, b1, (void*)h1);
    gemm_mlp_kernel<false, false><<<NN / 128, 256, 0, stream>>>(h1,    (const char*)Wp2, b2, (void*)h2);

    bn_stats1_kernel<<<128, 256, 0, stream>>>(h2, part);
    bn_stats2_kernel<<<1, 256, 0, stream>>>(part, gamma, beta, ss);
    bn_apply_kernel<<<NN * DIM / 4 / 256, 256, 0, stream>>>(h2, ss, (float*)d_out);
}
